// get_model_76201309765876
// MI455X (gfx1250) — matrix operand order repaired and checked by reading, not compiled, not run on hardware
//
#include <hip/hip_runtime.h>

typedef __attribute__((ext_vector_type(16))) _Float16 v16h;
typedef __attribute__((ext_vector_type(8)))  float    v8f;

#define BNEPS 1e-5f

// ---------------------------------------------------------------------------
// Transpose input (B,9,N) -> p0 (B,N,9) and x0 (B,N,3)
// ---------------------------------------------------------------------------
__global__ void k_transpose_in(const float* __restrict__ xyz,
                               float* __restrict__ p0, float* __restrict__ x0, int N) {
  int b = blockIdx.y;
  int n = blockIdx.x * blockDim.x + threadIdx.x;
  if (n >= N) return;
  #pragma unroll
  for (int c = 0; c < 9; ++c) {
    float v = xyz[((size_t)b * 9 + c) * N + n];
    p0[((size_t)b * N + n) * 9 + c] = v;
    if (c < 3) x0[((size_t)b * N + n) * 3 + c] = v;
  }
}

// ---------------------------------------------------------------------------
// Farthest point sampling: one workgroup per batch, dists register-resident.
// Wave32 shuffle argmax; 2 barriers per round instead of 9.
// ---------------------------------------------------------------------------
__global__ void k_fps(const float* __restrict__ xyz, int n, int npoint,
                      int* __restrict__ out) {
  const int b = blockIdx.x;
  const int t = threadIdx.x;
  const int T = 256;
  __shared__ float sv[8];
  __shared__ int   si[8];
  __shared__ float cpt[3];
  __shared__ int   sfar;
  float dist[32];
  #pragma unroll
  for (int jj = 0; jj < 32; ++jj) dist[jj] = 1e10f;
  if (t == 0) sfar = 0;
  __syncthreads();
  for (int it = 0; it < npoint; ++it) {
    int far = sfar;
    if (t == 0) {
      out[b * npoint + it] = far;
      cpt[0] = xyz[((size_t)b * n + far) * 3 + 0];
      cpt[1] = xyz[((size_t)b * n + far) * 3 + 1];
      cpt[2] = xyz[((size_t)b * n + far) * 3 + 2];
    }
    __syncthreads();
    float cx = cpt[0], cy = cpt[1], cz = cpt[2];
    float best = -1.f; int bi = 0;
    #pragma unroll
    for (int jj = 0; jj < 32; ++jj) {
      int j = t + jj * T;
      if (j < n) {
        const float* pp = &xyz[((size_t)b * n + j) * 3];
        float dx = pp[0] - cx, dy = pp[1] - cy, dz = pp[2] - cz;
        float d = dx * dx + dy * dy + dz * dz;
        float m = fminf(dist[jj], d);
        dist[jj] = m;
        if (m > best) { best = m; bi = j; }
      }
    }
    // intra-wave argmax (no barriers)
    #pragma unroll
    for (int o = 16; o > 0; o >>= 1) {
      float ov = __shfl_xor(best, o, 32);
      int   oi = __shfl_xor(bi, o, 32);
      if (ov > best || (ov == best && oi < bi)) { best = ov; bi = oi; }
    }
    if ((t & 31) == 0) { sv[t >> 5] = best; si[t >> 5] = bi; }
    __syncthreads();
    if (t < 32) {
      float bv = (t < 8) ? sv[t] : -1.f;
      int  bii = (t < 8) ? si[t] : 0;
      #pragma unroll
      for (int o = 4; o > 0; o >>= 1) {
        float ov = __shfl_xor(bv, o, 32);
        int   oi = __shfl_xor(bii, o, 32);
        if (ov > bv || (ov == bv && oi < bii)) { bv = ov; bii = oi; }
      }
      if (t == 0) sfar = bii;
    }
    __syncthreads();
  }
}

// ---------------------------------------------------------------------------
// Gather sampled centroids
// ---------------------------------------------------------------------------
__global__ void k_gather3(const float* __restrict__ xyz, const int* __restrict__ idx,
                          int n, int S, float* __restrict__ out) {
  int b = blockIdx.y;
  int s = blockIdx.x * blockDim.x + threadIdx.x;
  if (s >= S) return;
  int id = idx[b * S + s];
  #pragma unroll
  for (int c = 0; c < 3; ++c)
    out[((size_t)b * S + s) * 3 + c] = xyz[((size_t)b * n + id) * 3 + c];
}

// ---------------------------------------------------------------------------
// Ball query + grouped feature build. One wave32 per (b,s). Ordered first-k
// selection via wave ballot prefix counts. f layout: (B*S, k, cin+3).
// ---------------------------------------------------------------------------
__global__ void k_group(const float* __restrict__ xyz, const float* __restrict__ pts,
                        const float* __restrict__ new_xyz, float r2, int k, int n,
                        int S, int cin, float* __restrict__ f) {
  const int b = blockIdx.y, s = blockIdx.x, lane = threadIdx.x;
  __shared__ int idxs[64];
  const float qx = new_xyz[((size_t)b * S + s) * 3 + 0];
  const float qy = new_xyz[((size_t)b * S + s) * 3 + 1];
  const float qz = new_xyz[((size_t)b * S + s) * 3 + 2];
  int count = 0;
  for (int j0 = 0; j0 < n && count < k; j0 += 32) {
    int j = j0 + lane;
    bool pred = false;
    if (j < n) {
      const float* pp = &xyz[((size_t)b * n + j) * 3];
      float dx = pp[0] - qx, dy = pp[1] - qy, dz = pp[2] - qz;
      pred = (dx * dx + dy * dy + dz * dz) <= r2;
    }
    unsigned mask = __builtin_amdgcn_ballot_w32(pred);
    unsigned lt = ((unsigned)1 << lane) - 1u;
    int pre = __builtin_popcount(mask & lt);
    if (pred && count + pre < k) idxs[count + pre] = j;
    count += __builtin_popcount(mask);
  }
  __syncthreads();
  if (count > k) count = k;
  if (count == 0) { if (lane == 0) idxs[0] = 0; count = 1; }
  __syncthreads();
  int first = idxs[0];
  for (int slot = count + lane; slot < k; slot += 32) idxs[slot] = first;
  __syncthreads();
  for (int slot = 0; slot < k; ++slot) {
    int id = idxs[slot];
    size_t srcp = ((size_t)b * n + id) * cin;
    size_t dst = (((size_t)(b * S + s)) * k + slot) * (size_t)(cin + 3);
    for (int c = lane; c < cin; c += 32) f[dst + c] = pts[srcp + c];
    if (lane < 3)
      f[dst + cin + lane] = xyz[((size_t)b * n + id) * 3 + lane] -
                            new_xyz[((size_t)b * S + s) * 3 + lane];
  }
}

// ---------------------------------------------------------------------------
// WMMA GEMM: Y[M,Nc] = X[M,K] * W[Nc,K]^T  (fp32 memory, f16 WMMA, f32 acc)
// block = 128 threads (4 waves); block tile 64(M) x 16(N); K tiled by 32.
// Requires M % 64 == 0, Nc % 16 == 0 (M always is; the Nc = 196 layer is not); K zero-padded.
// Staging: unconditional clamped-address batched loads into registers,
// one-deep software pipeline (prefetch next K-tile during WMMA).
// Full tiles take the straight-line store path; only the remainder tile masks.
// ---------------------------------------------------------------------------
__global__ void k_gemm_wmma(const float* __restrict__ X, const float* __restrict__ W,
                            float* __restrict__ Y, int M, int K, int Nc) {
  __shared__ _Float16 As[64][34];
  __shared__ _Float16 Bs[16][34];
  const int t = threadIdx.x;
  const int wave = t >> 5, lane = t & 31, hl = lane >> 4, lm = lane & 15;
  const int m0 = blockIdx.x * 64, n0 = blockIdx.y * 16;
  const int ra = t >> 5;   // row phase (0..3); covers rows ra+4i
  const int ca = t & 31;   // K column within tile (lane-consecutive -> coalesced)
  v8f acc = {};
  float va[16];
  float vb[4];

  auto load_tile = [&](int k0) {
    int kk = k0 + ca;
    int kc = kk < K ? kk : K - 1;            // clamped safe address, no branches
    #pragma unroll
    for (int i = 0; i < 16; ++i)
      va[i] = X[(size_t)(m0 + ra + i * 4) * K + kc];
    #pragma unroll
    for (int i = 0; i < 4; ++i)
      vb[i] = W[(size_t)(n0 + ra + i * 4) * K + kc];
  };
  auto store_tile = [&](int k0) {
    if (k0 + 32 <= K) {                      // uniform: full tile, no masking
      #pragma unroll
      for (int i = 0; i < 16; ++i)
        As[ra + i * 4][ca] = (_Float16)va[i];
      #pragma unroll
      for (int i = 0; i < 4; ++i)
        Bs[ra + i * 4][ca] = (_Float16)vb[i];
    } else {                                 // remainder tile: 0/1 mask multiply
      float mk = (k0 + ca) < K ? 1.f : 0.f;
      #pragma unroll
      for (int i = 0; i < 16; ++i)
        As[ra + i * 4][ca] = (_Float16)(va[i] * mk);
      #pragma unroll
      for (int i = 0; i < 4; ++i)
        Bs[ra + i * 4][ca] = (_Float16)(vb[i] * mk);
    }
  };

  load_tile(0);
  for (int k0 = 0; k0 < K; k0 += 32) {
    store_tile(k0);
    __syncthreads();
    if (k0 + 32 < K) load_tile(k0 + 32);     // overlap next-tile loads with WMMA
    v16h af, bf;
    const int mrow = wave * 16 + lm;
    #pragma unroll
    for (int v = 0; v < 8; ++v) {
      int ka = ((v < 4) ? 0 : 16) + hl * 8 + 2 * (v & 3);
      af[2 * v]     = As[mrow][ka];
      af[2 * v + 1] = As[mrow][ka + 1];
      int kb = ((v < 4) ? 0 : 16) + hl * 8 + 2 * (v & 3);
      bf[2 * v]     = Bs[lm][kb];
      bf[2 * v + 1] = Bs[lm][kb + 1];
    }
    acc = __builtin_amdgcn_wmma_f32_16x16x32_f16(false, af, false, bf,
                                                 (short)0, acc, false, false);
    __syncthreads();
  }
  #pragma unroll
  for (int i = 0; i < 8; ++i) {
    int m = m0 + wave * 16 + hl * 8 + i;
    Y[(size_t)m * Nc + n0 + lm] = acc[i];
  }
}

// ---------------------------------------------------------------------------
// BatchNorm stats, coalesced two-stage:
//   k_zero -> k_bn_partial (16 chans x 16 row-walkers, atomic partials)
//   -> k_bn_finalize (fused scale/shift)
// ---------------------------------------------------------------------------
__global__ void k_zero(float* __restrict__ a, int n) {
  int i = blockIdx.x * blockDim.x + threadIdx.x;
  if (i < n) a[i] = 0.f;
}

__global__ void k_bn_partial(const float* __restrict__ Y, int R, int C,
                             float* __restrict__ sums /* [C] sum, [C] sumsq */) {
  const int tx = threadIdx.x & 15, ty = threadIdx.x >> 4;
  const int c = blockIdx.x * 16 + tx;
  const bool cok = c < C;
  int chunk = (R + gridDim.y - 1) / gridDim.y;
  int r0 = blockIdx.y * chunk;
  int r1 = R < r0 + chunk ? R : r0 + chunk;
  float s = 0.f, q = 0.f;
  for (int r = r0 + ty; r < r1; r += 16) {
    float v = cok ? Y[(size_t)r * C + c] : 0.f;   // lane-consecutive addresses
    s += v; q += v * v;
  }
  __shared__ float ls[16][17], lq[16][17];
  ls[ty][tx] = s; lq[ty][tx] = q;
  __syncthreads();
  if (ty == 0) {
    #pragma unroll
    for (int i = 1; i < 16; ++i) { s += ls[i][tx]; q += lq[i][tx]; }
    if (cok) { atomicAdd(&sums[c], s); atomicAdd(&sums[C + c], q); }
  }
}

__global__ void k_bn_finalize(const float* __restrict__ sums, int R, int C,
                              const float* __restrict__ g, const float* __restrict__ bt,
                              float* __restrict__ scale, float* __restrict__ shift) {
  for (int c = threadIdx.x; c < C; c += 256) {
    float mean = sums[c] / (float)R;
    float var = sums[C + c] / (float)R - mean * mean;
    float sc = g[c] * rsqrtf(var + BNEPS);
    scale[c] = sc;
    shift[c] = bt[c] - mean * sc;
  }
}

// Vectorized BN+ReLU apply (C % 4 == 0 for every BN'd layer)
__global__ void k_bn_relu4(float4* __restrict__ Y, const float* __restrict__ scale,
                           const float* __restrict__ shift, size_t total4, int C) {
  size_t i = blockIdx.x * (size_t)blockDim.x + threadIdx.x;
  if (i >= total4) return;
  int c = (int)((i * 4) % C);
  float4 v = Y[i];
  v.x = fmaxf(v.x * scale[c + 0] + shift[c + 0], 0.f);
  v.y = fmaxf(v.y * scale[c + 1] + shift[c + 1], 0.f);
  v.z = fmaxf(v.z * scale[c + 2] + shift[c + 2], 0.f);
  v.w = fmaxf(v.w * scale[c + 3] + shift[c + 3], 0.f);
  Y[i] = v;
}

// ---------------------------------------------------------------------------
// Max-pool over neighborhood k; write into concat buffer with channel offset
// ---------------------------------------------------------------------------
__global__ void k_maxpool(const float* __restrict__ F, int k, int C,
                          int Cstride, int coff, float* __restrict__ out, int rows) {
  size_t i = blockIdx.x * (size_t)blockDim.x + threadIdx.x;
  size_t total = (size_t)rows * C;
  if (i >= total) return;
  int row = (int)(i / C), c = (int)(i % C);
  float m = -1e30f;
  for (int kk = 0; kk < k; ++kk) {
    float v = F[((size_t)row * k + kk) * C + c];
    m = fmaxf(m, v);
  }
  out[(size_t)row * Cstride + coff + c] = m;
}

// ---------------------------------------------------------------------------
// 3-NN search + inverse-distance weights
// ---------------------------------------------------------------------------
__global__ void k_three_nn(const float* __restrict__ xyz1, const float* __restrict__ xyz2,
                           int n1, int n2, int* __restrict__ idx3, float* __restrict__ w3) {
  int b = blockIdx.y;
  int i = blockIdx.x * blockDim.x + threadIdx.x;
  if (i >= n1) return;
  float px = xyz1[((size_t)b * n1 + i) * 3 + 0];
  float py = xyz1[((size_t)b * n1 + i) * 3 + 1];
  float pz = xyz1[((size_t)b * n1 + i) * 3 + 2];
  float d0 = 1e30f, d1 = 1e30f, d2 = 1e30f;
  int i0 = 0, i1 = 0, i2 = 0;
  for (int j = 0; j < n2; ++j) {
    float dx = xyz2[((size_t)b * n2 + j) * 3 + 0] - px;
    float dy = xyz2[((size_t)b * n2 + j) * 3 + 1] - py;
    float dz = xyz2[((size_t)b * n2 + j) * 3 + 2] - pz;
    float d = dx * dx + dy * dy + dz * dz;
    if (d < d0)      { d2 = d1; i2 = i1; d1 = d0; i1 = i0; d0 = d; i0 = j; }
    else if (d < d1) { d2 = d1; i2 = i1; d1 = d; i1 = j; }
    else if (d < d2) { d2 = d; i2 = j; }
  }
  float w0 = 1.f / (d0 + 1e-8f), w1 = 1.f / (d1 + 1e-8f), w2 = 1.f / (d2 + 1e-8f);
  float ws = w0 + w1 + w2;
  size_t o = ((size_t)b * n1 + i) * 3;
  idx3[o + 0] = i0; idx3[o + 1] = i1; idx3[o + 2] = i2;
  w3[o + 0] = w0 / ws; w3[o + 1] = w1 / ws; w3[o + 2] = w2 / ws;
}

// ---------------------------------------------------------------------------
// Interpolate p2 with 3-NN weights and concat with p1 -> out (B,n1,c1+c2)
// ---------------------------------------------------------------------------
__global__ void k_interp_concat(const float* __restrict__ p1, const float* __restrict__ p2,
                                const int* __restrict__ idx3, const float* __restrict__ w3,
                                int n1, int n2, int c1, int c2, float* __restrict__ out) {
  int b = blockIdx.y;
  int C = c1 + c2;
  size_t gid = blockIdx.x * (size_t)blockDim.x + threadIdx.x;
  if (gid >= (size_t)n1 * C) return;
  int i = (int)(gid / C), c = (int)(gid % C);
  float v;
  if (c < c1) {
    v = p1[((size_t)b * n1 + i) * c1 + c];
  } else {
    int cc = c - c1;
    size_t o = ((size_t)b * n1 + i) * 3;
    int i0 = idx3[o], i1 = idx3[o + 1], i2 = idx3[o + 2];
    v = w3[o]     * p2[((size_t)b * n2 + i0) * c2 + cc]
      + w3[o + 1] * p2[((size_t)b * n2 + i1) * c2 + cc]
      + w3[o + 2] * p2[((size_t)b * n2 + i2) * c2 + cc];
  }
  out[((size_t)b * n1 + i) * C + c] = v;
}

// ---------------------------------------------------------------------------
// (B,S,C) -> (B,C,S) transpose (for the p4 output)
// ---------------------------------------------------------------------------
__global__ void k_transpose_cs(const float* __restrict__ in, int S, int C,
                               float* __restrict__ out, size_t total) {
  size_t i = blockIdx.x * (size_t)blockDim.x + threadIdx.x;
  if (i >= total) return;
  int b = (int)(i / ((size_t)S * C));
  int rem = (int)(i % ((size_t)S * C));
  int s = rem / C, c = rem % C;
  out[((size_t)b * C + c) * S + s] = in[i];
}

// ---------------------------------------------------------------------------
// Head: conv2 (13x128) + bias, transposed logits + softmax probs.
// Stream feature row once; W2 accesses are lane-uniform -> scalar loads.
// ---------------------------------------------------------------------------
#define CLSK 13
__global__ void k_head(const float* __restrict__ h, const float* __restrict__ W2,
                       const float* __restrict__ b2, int N, int C,
                       float* __restrict__ logitsT, float* __restrict__ probs) {
  int b = blockIdx.y;
  int n = blockIdx.x * blockDim.x + threadIdx.x;
  if (n >= N) return;
  const float* hr = &h[((size_t)b * N + n) * C];
  float lg[CLSK];
  #pragma unroll
  for (int o = 0; o < CLSK; ++o) lg[o] = b2[o];
  for (int c = 0; c < C; ++c) {
    float hv = hr[c];
    #pragma unroll
    for (int o = 0; o < CLSK; ++o) lg[o] += hv * W2[o * C + c];
  }
  float mx = -1e30f;
  #pragma unroll
  for (int o = 0; o < CLSK; ++o) mx = fmaxf(mx, lg[o]);
  float se = 0.f;
  float ex[CLSK];
  #pragma unroll
  for (int o = 0; o < CLSK; ++o) { ex[o] = expf(lg[o] - mx); se += ex[o]; }
  float inv = 1.f / se;
  #pragma unroll
  for (int o = 0; o < CLSK; ++o) {
    logitsT[((size_t)b * CLSK + o) * N + n] = lg[o];
    probs[((size_t)b * N + n) * CLSK + o] = ex[o] * inv;
  }
}

// ---------------------------------------------------------------------------
// Host orchestration
// ---------------------------------------------------------------------------
extern "C" void kernel_launch(void* const* d_in, const int* in_sizes, int n_in,
                              void* d_out, int out_size, void* d_ws, size_t ws_size,
                              hipStream_t stream) {
  (void)in_sizes; (void)n_in; (void)out_size; (void)ws_size;
  const int B = 4, N = 8192, CLS = 13;
  static const int NCENT[4] = {1024, 256, 64, 16};
  static const int SA_INC[4] = {9, 96, 384, 768};
  static const float RAD2[4][2] = {{0.0025f, 0.01f}, {0.01f, 0.04f},
                                   {0.04f, 0.16f}, {0.16f, 0.64f}};
  static const int NSAMP[2] = {16, 32};
  static const int SA_MLPS[4][2][3] = {
      {{16, 16, 32}, {32, 32, 64}},
      {{64, 64, 128}, {128, 128, 256}},
      {{128, 196, 256}, {256, 256, 512}},
      {{256, 256, 512}, {512, 512, 1024}}};
  static const int FP_MLPS[4][2] = {{256, 512}, {256, 512}, {256, 128}, {128, 128}};
  static const int FP_IN[4] = {2304, 896, 608, 128};

  struct Layer { const float *W, *b, *g, *bt; int cin, cout; };

  int p = 0;
  const float* xyz = (const float*)d_in[p++];
  Layer sa[4][2][3];
  for (int i = 0; i < 4; ++i)
    for (int br = 0; br < 2; ++br) {
      int cin = SA_INC[i] + 3;
      for (int l = 0; l < 3; ++l) {
        Layer L;
        L.W = (const float*)d_in[p++]; L.b = (const float*)d_in[p++];
        L.g = (const float*)d_in[p++]; L.bt = (const float*)d_in[p++];
        L.cin = cin; L.cout = SA_MLPS[i][br][l]; cin = L.cout;
        sa[i][br][l] = L;
      }
    }
  Layer fpl[4][2];
  for (int i = 0; i < 4; ++i) {
    int cin = FP_IN[i];
    for (int l = 0; l < 2; ++l) {
      Layer L;
      L.W = (const float*)d_in[p++]; L.b = (const float*)d_in[p++];
      L.g = (const float*)d_in[p++]; L.bt = (const float*)d_in[p++];
      L.cin = cin; L.cout = FP_MLPS[i][l]; cin = L.cout;
      fpl[i][l] = L;
    }
  }
  Layer conv1;
  conv1.W = (const float*)d_in[p++]; conv1.b = (const float*)d_in[p++];
  conv1.g = (const float*)d_in[p++]; conv1.bt = (const float*)d_in[p++];
  conv1.cin = 128; conv1.cout = 128;
  const float* W2 = (const float*)d_in[p++];
  const float* b2 = (const float*)d_in[p++];

  // ---- workspace arena ----
  char* base = (char*)d_ws;
  size_t off = 0;
  auto allocf = [&](size_t elems) -> float* {
    float* r = (float*)(base + off);
    off += ((elems * 4 + 255) & ~(size_t)255);
    return r;
  };
  auto alloci = [&](size_t elems) -> int* {
    int* r = (int*)(base + off);
    off += ((elems * 4 + 255) & ~(size_t)255);
    return r;
  };

  float* xs[5]; float* ps[5]; int PC[5];
  ps[0] = allocf((size_t)B * N * 9);
  xs[0] = allocf((size_t)B * N * 3);
  PC[0] = 9;
  for (int i = 0; i < 4; ++i) {
    int S = NCENT[i];
    int C = SA_MLPS[i][0][2] + SA_MLPS[i][1][2];
    xs[i + 1] = allocf((size_t)B * S * 3);
    ps[i + 1] = allocf((size_t)B * S * C);
    PC[i + 1] = C;
  }
  float* fr3 = allocf((size_t)B * 64 * 512);
  float* fr2 = allocf((size_t)B * 256 * 512);
  float* fr1 = allocf((size_t)B * 1024 * 128);
  float* scale = allocf(1024);
  float* shift = allocf(1024);
  float* sums  = allocf(2048);
  int* fi = alloci((size_t)B * 1024);
  int* idx3 = alloci((size_t)B * N * 3);
  float* w3 = allocf((size_t)B * N * 3);
  float* G  = allocf(4500000);
  float* TA = allocf(9000000);
  float* TB = allocf(9000000);

  auto gemm = [&](const float* X, const float* W, float* Y, int M, int K, int Nc) {
    dim3 g(M / 64, Nc / 16);
    k_gemm_wmma<<<g, 128, 0, stream>>>(X, W, Y, M, K, Nc);
  };
  auto bnrelu = [&](float* Y, const Layer& L, int R) {
    int C = L.cout;
    k_zero<<<(2 * C + 255) / 256, 256, 0, stream>>>(sums, 2 * C);
    int rsplit = R >= 16384 ? 64 : (R >= 2048 ? 16 : 4);
    k_bn_partial<<<dim3((C + 15) / 16, rsplit), 256, 0, stream>>>(Y, R, C, sums);
    k_bn_finalize<<<1, 256, 0, stream>>>(sums, R, C, L.g, L.bt, scale, shift);
    size_t tot4 = ((size_t)R * C) / 4;
    k_bn_relu4<<<(unsigned)((tot4 + 255) / 256), 256, 0, stream>>>(
        (float4*)Y, scale, shift, tot4, C);
  };

  // transpose input
  k_transpose_in<<<dim3((N + 255) / 256, B), 256, 0, stream>>>(xyz, ps[0], xs[0], N);

  // ---- SA stages ----
  int npts[5] = {N, NCENT[0], NCENT[1], NCENT[2], NCENT[3]};
  for (int lvl = 0; lvl < 4; ++lvl) {
    int n = npts[lvl], S = NCENT[lvl];
    k_fps<<<B, 256, 0, stream>>>(xs[lvl], n, S, fi);
    k_gather3<<<dim3((S + 63) / 64, B), 64, 0, stream>>>(xs[lvl], fi, n, S, xs[lvl + 1]);
    int coff = 0;
    for (int br = 0; br < 2; ++br) {
      int K = NSAMP[br];
      k_group<<<dim3(S, B), 32, 0, stream>>>(xs[lvl], ps[lvl], xs[lvl + 1],
                                             RAD2[lvl][br], K, n, S, PC[lvl], G);
      int M = B * S * K;
      float* cur = G;
      for (int l = 0; l < 3; ++l) {
        float* nxt = (l & 1) ? TB : TA;
        gemm(cur, sa[lvl][br][l].W, nxt, M, sa[lvl][br][l].cin, sa[lvl][br][l].cout);
        bnrelu(nxt, sa[lvl][br][l], M);
        cur = nxt;
      }
      int Cl = SA_MLPS[lvl][br][2];
      size_t tot = (size_t)B * S * Cl;
      k_maxpool<<<(unsigned)((tot + 255) / 256), 256, 0, stream>>>(
          cur, K, Cl, PC[lvl + 1], coff, ps[lvl + 1], B * S);
      coff += Cl;
    }
  }

  // ---- FP stages ----
  auto fp = [&](int lvl, const float* x1, const float* x2, const float* p1c,
                const float* p2c, int n1, int n2, int c1, int c2, float* outbuf) {
    k_three_nn<<<dim3((n1 + 127) / 128, B), 128, 0, stream>>>(x1, x2, n1, n2, idx3, w3);
    int C = c1 + c2;
    size_t tot = (size_t)n1 * C;
    k_interp_concat<<<dim3((unsigned)((tot + 255) / 256), B), 256, 0, stream>>>(
        p1c, p2c, idx3, w3, n1, n2, c1, c2, G);
    int M = B * n1;
    float* cur = G;
    for (int l = 0; l < 2; ++l) {
      float* nxt = (l & 1) ? TB : TA;
      gemm(cur, fpl[lvl][l].W, nxt, M, fpl[lvl][l].cin, fpl[lvl][l].cout);
      bnrelu(nxt, fpl[lvl][l], M);
      cur = nxt;
    }
    if (cur != outbuf)
      hipMemcpyAsync(outbuf, cur, (size_t)M * fpl[lvl][1].cout * 4,
                     hipMemcpyDeviceToDevice, stream);
  };

  // fp4: x3 <- x4
  fp(0, xs[3], xs[4], ps[3], ps[4], 64, 16, PC[3], PC[4], fr3);
  // fp3: x2 <- x3
  fp(1, xs[2], xs[3], ps[2], fr3, 256, 64, PC[2], 512, fr2);
  // fp2: x1 <- x2
  fp(2, xs[1], xs[2], ps[1], fr2, 1024, 256, PC[1], 512, fr1);
  // fp1: x0 <- x1 (no p1 skip features)
  fp(3, xs[0], xs[1], nullptr, fr1, N, 1024, 0, 128, TB);

  // ---- head ----
  gemm(TB, conv1.W, TA, B * N, 128, 128);
  bnrelu(TA, conv1, B * N);

  float* logitsT = (float*)d_out;
  float* p4T = logitsT + (size_t)B * CLS * N;
  float* probs = p4T + (size_t)B * 1536 * 16;
  k_head<<<dim3((N + 127) / 128, B), 128, 0, stream>>>(TA, W2, b2, N, 128, logitsT, probs);
  size_t p4tot = (size_t)B * 16 * 1536;
  k_transpose_cs<<<(unsigned)((p4tot + 255) / 256), 256, 0, stream>>>(ps[4], 16, 1536, p4T, p4tot);
}
